// SpatialGATAttention_8821862826026
// MI455X (gfx1250) — matrix operand order repaired and checked by reading, not compiled, not run on hardware
//
#include <hip/hip_runtime.h>
#include <hip/hip_bf16.h>

// Shapes from the reference
#define B_  8
#define C_  64
#define T_  24
#define N_  512
#define H_  2
#define D_  32
#define M_  (B_*T_)        // 192
#define NEG_    (-1e9f)
#define LN_EPS_ 1e-5f

typedef _Float16 half_t;
typedef __attribute__((ext_vector_type(16))) _Float16 v16h;
typedef __attribute__((ext_vector_type(8)))  float    v8f;
typedef __attribute__((ext_vector_type(4)))  unsigned int u32x4;
typedef __attribute__((ext_vector_type(4)))  int      i32x4;
typedef __attribute__((ext_vector_type(8)))  int      i32x8;

union F16Frag { v16h v; unsigned int u[8]; half_t h[16]; };

__device__ inline v8f wmma_f32_f16(v16h a, v16h b, v8f c) {
  // 8 args: (neg_a, A, neg_b, B, c_mod, C, reuse_a, reuse_b)
  return __builtin_amdgcn_wmma_f32_16x16x32_f16(false, a, false, b, (short)0, c,
                                                false, false);
}

// A fragment (16x32 f16, row-major source with leading dim ldm, K origin k0).
// ISA layout: lanes 0-15 rows 0-15 / K {0..7,16..23}; lanes 16-31 K {8..15,24..31}.
__device__ inline v16h load_a_frag(const half_t* A, int ldm, int k0, int lane) {
  const int row = lane & 15;
  const int kb  = (lane < 16) ? 0 : 8;
  const half_t* p = A + row * ldm + k0;
  F16Frag f;
#pragma unroll
  for (int i = 0; i < 4; ++i)
    f.u[i]     = *(const unsigned int*)(p + kb + 2 * i);
#pragma unroll
  for (int i = 0; i < 4; ++i)
    f.u[4 + i] = *(const unsigned int*)(p + 16 + kb + 2 * i);
  return f.v;
}

// B fragment (32x16) from fp32 weight, row-major W[d][64]; we need B[c,d]=W[d,c].
__device__ inline v16h load_b_frag_w(const float* W, int dcol, int k0, int lane) {
  const int kb = (lane < 16) ? 0 : 8;
  const float* p = W + dcol * 64 + k0 + kb;
  F16Frag f;
#pragma unroll
  for (int i = 0; i < 8; ++i) {
    const int o = 2 * i + (i >= 4 ? 8 : 0);
    f.h[2 * i]     = (half_t)p[o];
    f.h[2 * i + 1] = (half_t)p[o + 1];
  }
  return f.v;
}

// B fragment (32x16) from LDS f16 matrix V[512][64]; K rows j0.., column c.
__device__ inline v16h load_b_frag_lds(const half_t* V, int j0, int c, int lane) {
  const int kb = (lane < 16) ? 0 : 8;
  F16Frag f;
#pragma unroll
  for (int i = 0; i < 8; ++i) {
    const int o = kb + 2 * i + (i >= 4 ? 8 : 0);
    f.h[2 * i]     = V[(j0 + o)     * 64 + c];
    f.h[2 * i + 1] = V[(j0 + o + 1) * 64 + c];
  }
  return f.v;
}

// ---------------------------------------------------------------------------
// Kernel 1: fused QKV projection.  Q,K reduced to s_src/s_dst on the fly;
// only V (f16) is written out.  One WG = one m, 16 consecutive nodes.
// ---------------------------------------------------------------------------
__global__ __launch_bounds__(128) void gat_proj_kernel(
    const float* __restrict__ x,
    const float* __restrict__ Wq, const float* __restrict__ Wk,
    const float* __restrict__ Wv,
    const float* __restrict__ a_src, const float* __restrict__ a_dst,
    float* __restrict__ sSrc, float* __restrict__ sDst,
    half_t* __restrict__ Vh)
{
  __shared__ half_t Ash[16][64];   // x tile, f16
  __shared__ float  sS[2][16];
  __shared__ float  sD[2][16];

  const int tid  = threadIdx.x;
  const int wave = tid >> 5;
  const int lane = tid & 31;
  const int m  = blockIdx.x >> 5;          // 0..191
  const int n0 = (blockIdx.x & 31) << 4;   // 0..496
  const int b = m / T_, t = m % T_;

  // Stage x tile: Ash[rr][c] = x[b, c, t, n0+rr]  (coalesced along n)
  for (int e = tid; e < 16 * 64; e += 128) {
    const int rr = e & 15, c = e >> 4;
    Ash[rr][c] = (half_t)x[(((size_t)b * C_ + c) * T_ + t) * N_ + n0 + rr];
  }
  if (tid < 32) { sS[tid >> 4][tid & 15] = 0.f; sD[tid >> 4][tid & 15] = 0.f; }
  __syncthreads();

  const int col = lane & 15;
  const int cc  = wave * 16 + col;       // output channel 0..63
  const int h   = cc >> 5, d = cc & 31;

  v8f accQ = {0,0,0,0,0,0,0,0};
  v8f accK = {0,0,0,0,0,0,0,0};
  v8f accV = {0,0,0,0,0,0,0,0};
#pragma unroll
  for (int k0 = 0; k0 < 64; k0 += 32) {
    v16h a = load_a_frag(&Ash[0][0], 64, k0, lane);
    accQ = wmma_f32_f16(a, load_b_frag_w(Wq, cc, k0, lane), accQ);
    accK = wmma_f32_f16(a, load_b_frag_w(Wk, cc, k0, lane), accK);
    accV = wmma_f32_f16(a, load_b_frag_w(Wv, cc, k0, lane), accV);
  }

  const float as = a_src[h * D_ + d];
  const float ad = a_dst[h * D_ + d];
  const int rbase = (lane < 16) ? 0 : 8;
#pragma unroll
  for (int r = 0; r < 8; ++r) {
    const int row = rbase + r;
    const int n   = n0 + row;
    Vh[((size_t)m * N_ + n) * C_ + cc] = (half_t)accV[r];
    atomicAdd(&sS[h][row], accQ[r] * as);   // ds_add_f32
    atomicAdd(&sD[h][row], accK[r] * ad);
  }
  __syncthreads();

  if (tid < 32) {
    sSrc[((size_t)m * H_ + (tid >> 4)) * N_ + n0 + (tid & 15)] =
        sS[tid >> 4][tid & 15];
  } else if (tid < 64) {
    const int q = tid - 32;
    sDst[((size_t)m * H_ + (q >> 4)) * N_ + n0 + (q & 15)] =
        sD[q >> 4][q & 15];
  }
}

// ---------------------------------------------------------------------------
// Kernel 2: scores + softmax + P@V (WMMA) + LayerNorm + output transpose.
// One WG = one m, 16 destination nodes, both heads.
// V slab (512x64 f16, 64KB contiguous) is DMA'd into LDS by the Tensor Data
// Mover at kernel entry, overlapping the whole score/softmax phase; a single
// s_wait_tensorcnt + barrier gates the WMMA GEMM loop (no mid-loop barriers).
// ---------------------------------------------------------------------------
// Dynamic LDS layout (bytes):
#define SM_VFULL   0                          // 512*64*2  = 65536
#define SM_PN      65536                      // 2*16*512*2 = 32768
#define SM_SDLOC   (SM_PN + 32768)            // 2*512*4   = 4096
#define SM_OUTSH   (SM_SDLOC + 4096)          // 16*64*4   = 4096
#define SM_SSLOC   (SM_OUTSH + 4096)          // 32*4
#define SM_ROWSUM  (SM_SSLOC + 128)           // 32*4
#define SM_ROWMU   (SM_ROWSUM + 128)          // 16*4
#define SM_ROWRS   (SM_ROWMU + 64)            // 16*4
#define SM_TOTAL   (SM_ROWRS + 64)            // = 106880

__global__ __launch_bounds__(128) void gat_attn_kernel(
    const int* __restrict__ gso,
    const float* __restrict__ sSrc, const float* __restrict__ sDst,
    const half_t* __restrict__ Vh,
    const float* __restrict__ gamma, const float* __restrict__ beta,
    float* __restrict__ y)
{
  extern __shared__ char smem[];
  half_t* Vfull  = (half_t*)(smem + SM_VFULL);   // [512][64]
  half_t* Pn     = (half_t*)(smem + SM_PN);      // [2][16][512]
  float*  sdLoc  = (float*)(smem + SM_SDLOC);    // [2][512]
  float*  outSh  = (float*)(smem + SM_OUTSH);    // [16][64]
  float*  ssLoc  = (float*)(smem + SM_SSLOC);    // [2][16]
  float*  rowSum = (float*)(smem + SM_ROWSUM);   // [32]
  float*  rowMu  = (float*)(smem + SM_ROWMU);    // [16]
  float*  rowRs  = (float*)(smem + SM_ROWRS);    // [16]

  const int tid  = threadIdx.x;
  const int wave = tid >> 5;
  const int lane = tid & 31;
  const int m  = blockIdx.x >> 5;
  const int i0 = (blockIdx.x & 31) << 4;
  const int b = m / T_, t = m % T_;

  // --- Issue TDM: Vfull[512][64] <- Vh[m,:,:]  (64KB, 8192 x 8-byte elems) ---
#if __has_builtin(__builtin_amdgcn_tensor_load_to_lds)
  if (wave == 0) {
    const unsigned long long ga =
        (unsigned long long)(const void*)(Vh + (size_t)m * N_ * C_);
    const unsigned int ldsOff = (unsigned int)(size_t)(void*)Vfull;
    u32x4 g0;
    g0[0] = 1u;                                      // count=1, user mode
    g0[1] = ldsOff;                                  // lds_addr
    g0[2] = (unsigned int)(ga & 0xFFFFFFFFull);      // global_addr[31:0]
    g0[3] = (unsigned int)((ga >> 32) & 0x01FFFFFFu) // global_addr[56:32]
            | 0x80000000u;                           // type=2 (bits 127:126)
    i32x8 g1;
    g1[0] = (3 << 16);        // workgroup_mask=0, data_size=3 (8B)
    g1[1] = (int)(8192u << 16);   // tensor_dim0[15:0] in bits 63:48
    g1[2] = (int)(1u << 16);      // tensor_dim0 hi=0 | tensor_dim1=1
    g1[3] = (int)(8192u << 16);   // tensor_dim1 hi=0 | tile_dim0=8192
    g1[4] = 0;                    // tile_dim1=0 (unused), tile_dim2=0
    g1[5] = 8192;                 // tensor_dim0_stride lo
    g1[6] = 0;                    // stride hi | tensor_dim1_stride lo
    g1[7] = 0;
    i32x4 z4 = {0, 0, 0, 0};
#if __has_include(<hip/amd_detail/amd_gfx1250_TDM.h>)
    i32x8 z8 = {0, 0, 0, 0, 0, 0, 0, 0};
    __builtin_amdgcn_tensor_load_to_lds(g0, g1, z4, z4, z8, 0);   // clang-23 form
#else
    __builtin_amdgcn_tensor_load_to_lds(g0, g1, z4, z4, 0);       // ROCm 7.2 form
#endif
  }
#define GAT_TDM 1
#endif

  for (int e = tid; e < H_ * N_; e += 128)
    sdLoc[e] = sDst[(size_t)m * H_ * N_ + e];
  if (tid < 32)
    ssLoc[tid] = sSrc[((size_t)m * H_ + (tid >> 4)) * N_ + i0 + (tid & 15)];
  __syncthreads();

  // Scores: row r = h*16+i handled by 4 lanes (q = tid&3), 128 j each.
  const int r = tid >> 2, q = tid & 3;
  const int h = r >> 4, i = r & 15;
  const float si = ssLoc[h * 16 + i];
  const int* grow = gso + (size_t)(i0 + i) * N_;
  __builtin_prefetch(grow + q * 128, 0, 1);   // global_prefetch_b8

  float mx = -3.0e38f;
  for (int j = q * 128; j < q * 128 + 128; ++j) {
    float s = si + sdLoc[h * N_ + j];
    s = (s < 0.f) ? 0.2f * s : s;          // leaky_relu(0.2) BEFORE mask
    if (grow[j] == 0) s += NEG_;
    mx = fmaxf(mx, s);
  }
  mx = fmaxf(mx, __shfl_xor(mx, 1));
  mx = fmaxf(mx, __shfl_xor(mx, 2));

  float lsum = 0.f;
  for (int j = q * 128; j < q * 128 + 128; ++j) {
    float s = si + sdLoc[h * N_ + j];
    s = (s < 0.f) ? 0.2f * s : s;
    if (grow[j] == 0) s += NEG_;
    const float e = __expf(s - mx);
    lsum += e;
    Pn[(h * 16 + i) * N_ + j] = (half_t)e;
  }
  lsum += __shfl_xor(lsum, 1);
  lsum += __shfl_xor(lsum, 2);
  if (q == 0) rowSum[r] = lsum;

  // Gate the GEMM on the TDM transfer (issuing wave waits, then all sync).
#ifdef GAT_TDM
  if (wave == 0) __builtin_amdgcn_s_wait_tensorcnt(0);
  __syncthreads();
#else
  __syncthreads();
  {  // fallback: cooperative dword copy of the V slab
    const unsigned int* src =
        (const unsigned int*)(Vh + (size_t)m * N_ * C_);
    unsigned int* dst = (unsigned int*)Vfull;
    for (int e = tid; e < N_ * C_ / 2; e += 128) dst[e] = src[e];
    __syncthreads();
  }
#endif

  // GEMM: out[h, i, d] = sum_j Pn[h][i][j] * V[m, j, h, d]; wave -> (h, dtile)
  const int gh    = wave >> 1;
  const int dtile = wave & 1;
  const int col   = lane & 15;
  const int bcol  = gh * 32 + dtile * 16 + col;
  const half_t* Pbase = Pn + gh * 16 * N_;
  v8f acc = {0,0,0,0,0,0,0,0};
#pragma unroll 4
  for (int kt = 0; kt < 16; ++kt) {
    v16h a  = load_a_frag(Pbase, N_, kt * 32, lane);
    v16h bb = load_b_frag_lds(Vfull, kt * 32, bcol, lane);
    acc = wmma_f32_f16(a, bb, acc);
  }
  __syncthreads();

  const int rbase = (lane < 16) ? 0 : 8;
#pragma unroll
  for (int rr = 0; rr < 8; ++rr) {
    const int row = rbase + rr;
    outSh[row * 64 + wave * 16 + col] = acc[rr] * (1.f / rowSum[gh * 16 + row]);
  }
  __syncthreads();

  // LayerNorm stats per row over C=64; 8 lanes per row, 8 channels each.
  const int lrow = tid >> 3, l8 = tid & 7;
  float s1 = 0.f, s2 = 0.f;
#pragma unroll
  for (int k = 0; k < 8; ++k) {
    const float vv = outSh[lrow * 64 + l8 * 8 + k];
    s1 += vv; s2 += vv * vv;
  }
  s1 += __shfl_xor(s1, 1); s2 += __shfl_xor(s2, 1);
  s1 += __shfl_xor(s1, 2); s2 += __shfl_xor(s2, 2);
  s1 += __shfl_xor(s1, 4); s2 += __shfl_xor(s2, 4);
  if (l8 == 0) {
    const float mean = s1 * (1.f / 64.f);
    const float var  = s2 * (1.f / 64.f) - mean * mean;
    rowMu[lrow] = mean;
    rowRs[lrow] = rsqrtf(var + LN_EPS_);
  }
  __syncthreads();

  // Write-out transposed for coalescing: 16 consecutive lanes -> consecutive n.
  for (int e = tid; e < 16 * 64; e += 128) {
    const int row = e & 15, c = e >> 4;
    const float vv = (outSh[row * 64 + c] - rowMu[row]) * rowRs[row] * gamma[c]
                     + beta[c];
    y[(((size_t)b * C_ + c) * T_ + t) * N_ + i0 + row] = vv;
  }
}

// ---------------------------------------------------------------------------
extern "C" void kernel_launch(void* const* d_in, const int* in_sizes, int n_in,
                              void* d_out, int out_size, void* d_ws,
                              size_t ws_size, hipStream_t stream) {
  const float* x     = (const float*)d_in[0];
  const int*   gso   = (const int*)  d_in[1];
  const float* Wq    = (const float*)d_in[2];
  const float* Wk    = (const float*)d_in[3];
  const float* Wv    = (const float*)d_in[4];
  const float* a_src = (const float*)d_in[5];
  const float* a_dst = (const float*)d_in[6];
  const float* gamma = (const float*)d_in[7];
  const float* beta  = (const float*)d_in[8];
  float* y = (float*)d_out;

  // Workspace: sSrc | sDst | V(f16)
  float*  ws   = (float*)d_ws;
  float*  sSrc = ws;                           // M*H*N floats
  float*  sDst = ws + (size_t)M_ * H_ * N_;    // M*H*N floats
  half_t* Vh   = (half_t*)(ws + 2 * (size_t)M_ * H_ * N_);  // M*N*C halves

  const int nblocks = M_ * (N_ / 16);          // 6144
  gat_proj_kernel<<<nblocks, 128, 0, stream>>>(x, Wq, Wk, Wv, a_src, a_dst,
                                               sSrc, sDst, Vh);
  gat_attn_kernel<<<nblocks, 128, SM_TOTAL, stream>>>(gso, sSrc, sDst, Vh,
                                                      gamma, beta, y);
}
